// EdgeWeightFromDistance_558345749245
// MI455X (gfx1250) — compile-verified, matrix operand order checked by reading, not run on hardware
//
#include <hip/hip_runtime.h>

#define NODE_DIM 64
#define HID      128
#define IN_DIM   129           // 64 + 64 + 1
#define KPAD     160           // IN_DIM padded to 5 K-tiles of 32
#define NKT      5             // K tiles (K=32 each)
#define NNT      8             // N tiles (N=16 each), 128/16
#define NFRAG    (NKT * NNT)   // 40 B fragments
#define TILE_E   16            // edges per wave tile (WMMA M)
#define WAVES_PB 4
#define BLOCK_E  (TILE_E * WAVES_PB)

typedef __attribute__((ext_vector_type(16))) _Float16 v16h;
typedef __attribute__((ext_vector_type(8)))  _Float16 v8h;
typedef __attribute__((ext_vector_type(8)))  float    v8f;
typedef __attribute__((ext_vector_type(4)))  float    v4f;

// ---------------------------------------------------------------------------
// Pack W1 (f32 [129][128], row-major) into per-lane WMMA B-fragment layout,
// zero-padded to K=160, f16.  packed[frag*512 + lane*16 + j]:
//   frag = kt*8 + nt
//   lane<16 : j 0..7 -> K kt*32+{0..7},   j 8..15 -> K kt*32+{16..23}
//   lane>=16: j 0..7 -> K kt*32+{8..15},  j 8..15 -> K kt*32+{24..31}
//   column  n = nt*16 + (lane & 15)
// ---------------------------------------------------------------------------
__global__ void pack_w1_kernel(const float* __restrict__ W1,
                               _Float16* __restrict__ packed) {
    int t = blockIdx.x * blockDim.x + threadIdx.x;
    if (t >= NFRAG * 32 * 16) return;
    int j    = t & 15;
    int lane = (t >> 4) & 31;
    int frag = t >> 9;
    int kt = frag >> 3;
    int nt = frag & 7;
    int kk = ((lane >> 4) << 3) + ((j >> 3) << 4) + (j & 7);
    int K  = kt * 32 + kk;
    int n  = nt * 16 + (lane & 15);
    float v = (K < IN_DIM) ? W1[K * HID + n] : 0.0f;
    packed[t] = (_Float16)v;
}

// ---------------------------------------------------------------------------
// Main kernel: one 16-edge tile per wave, 4 waves per block.
// ---------------------------------------------------------------------------
__global__ void __launch_bounds__(128)
edge_mlp_kernel(const float* __restrict__ x,  const float* __restrict__ y,
                const int*   __restrict__ ei, const float* __restrict__ posx,
                const float* __restrict__ posy,
                const _Float16* __restrict__ packedW1,
                const float* __restrict__ b1, const float* __restrict__ W2,
                const float* __restrict__ b2, float* __restrict__ out, int E) {
    __shared__ _Float16 sW1[NFRAG * 512];                 // 40 KB packed W1
    __shared__ _Float16 tile[WAVES_PB][TILE_E][KPAD];     // 20 KB A tiles

    const int tid  = threadIdx.x;
    const int lane = tid & 31;
    const int wave = tid >> 5;

    // ---- stage packed W1 into LDS (cooperative, 16B chunks) ----
    for (int i = tid; i < NFRAG * 512 / 8; i += 128)
        ((v8h*)sW1)[i] = ((const v8h*)packedW1)[i];

    // ---- stage A tile: lane = (row r, half hlf); hlf0 -> x[src], hlf1 -> y[dst]
    const int ebase = (blockIdx.x * WAVES_PB + wave) * TILE_E;
    const int r   = lane & 15;
    const int hlf = lane >> 4;
    const int e   = ebase + r;
    const int ec  = (e < E) ? e : (E - 1);
    const int node = hlf ? ei[E + ec] : ei[ec];

    // zero pad columns 128..159 of my row (hlf0: 128..143, hlf1: 144..159)
    {
        v8h z = {};
        _Float16* zp = &tile[wave][r][128 + hlf * 16];
        *(v8h*)zp       = z;
        *(v8h*)(zp + 8) = z;
    }

    // gather 64 node features, convert f32->f16, write row-major into tile
    {
        const float* srow = (hlf ? y : x) + (size_t)node * NODE_DIM;
        _Float16*    drow = &tile[wave][r][hlf * NODE_DIM];
        #pragma unroll
        for (int jj = 0; jj < NODE_DIM; jj += 8) {
            v4f f0 = *(const v4f*)(srow + jj);
            v4f f1 = *(const v4f*)(srow + jj + 4);
            v8h h;
            h[0] = (_Float16)f0[0]; h[1] = (_Float16)f0[1];
            h[2] = (_Float16)f0[2]; h[3] = (_Float16)f0[3];
            h[4] = (_Float16)f1[0]; h[5] = (_Float16)f1[1];
            h[6] = (_Float16)f1[2]; h[7] = (_Float16)f1[3];
            *(v8h*)(drow + jj) = h;
        }
    }

    // distance: hlf0 holds pos_x[src], hlf1 holds pos_y[dst]; swap via shfl_xor
    {
        const float* prow = (hlf ? posy : posx) + (size_t)node * 3;
        float p0 = prow[0], p1 = prow[1], p2 = prow[2];
        float q0 = __shfl_xor(p0, 16, 32);
        float q1 = __shfl_xor(p1, 16, 32);
        float q2 = __shfl_xor(p2, 16, 32);
        float d0 = p0 - q0, d1 = p1 - q1, d2 = p2 - q2;  // sign irrelevant
        float dist = sqrtf(d0 * d0 + d1 * d1 + d2 * d2);
        if (hlf == 0) tile[wave][r][128] = (_Float16)dist;
    }

    __syncthreads();

    // ---- layer 1: [16 x 160] @ [160 x 128] via 5x8 WMMA f16 -> f32 ----
    v8f C[NNT] = {};
    #pragma unroll
    for (int kt = 0; kt < NKT; ++kt) {
        // A fragment: lane<16 -> K {0..7,16..23}; lane>=16 -> K {8..15,24..31}
        const _Float16* arow = &tile[wave][r][kt * 32 + hlf * 8];
        v16h a;
        *(v8h*)&a       = *(const v8h*)arow;
        *((v8h*)&a + 1) = *(const v8h*)(arow + 16);
        #pragma unroll
        for (int nt = 0; nt < NNT; ++nt) {
            const _Float16* bp = &sW1[((kt * NNT + nt) * 32 + lane) * 16];
            v16h b;
            *(v8h*)&b       = *(const v8h*)bp;
            *((v8h*)&b + 1) = *(const v8h*)(bp + 8);
            C[nt] = __builtin_amdgcn_wmma_f32_16x16x32_f16(
                        false, a, false, b, (short)0, C[nt], false, false);
        }
    }

    // ---- bias + relu + layer 2 (dot with W2[128]) in VALU ----
    // C layout: VGPR rr, lanes 0-15 -> M=rr, lanes 16-31 -> M=8+rr; N = lane&15
    const int ln = lane & 15;
    float pacc[8] = {0.f, 0.f, 0.f, 0.f, 0.f, 0.f, 0.f, 0.f};
    #pragma unroll
    for (int nt = 0; nt < NNT; ++nt) {
        const int n = nt * 16 + ln;
        const float bb = b1[n];
        const float ww = W2[n];
        #pragma unroll
        for (int rr = 0; rr < 8; ++rr) {
            float h = C[nt][rr] + bb;
            h = (h > 0.f) ? h : 0.f;
            pacc[rr] = fmaf(h, ww, pacc[rr]);
        }
    }

    const float b2v = b2[0];
    #pragma unroll
    for (int rr = 0; rr < 8; ++rr) {
        float v = pacc[rr];
        v += __shfl_xor(v, 1, 16);     // reduce over the 16 N lanes
        v += __shfl_xor(v, 2, 16);
        v += __shfl_xor(v, 4, 16);
        v += __shfl_xor(v, 8, 16);
        v += b2v;
        v = (v > 0.f) ? v : 0.f;
        const int M  = hlf * 8 + rr;
        const int eo = ebase + M;
        if (ln == rr && eo < E) out[eo] = v;
    }
}

// ---------------------------------------------------------------------------
extern "C" void kernel_launch(void* const* d_in, const int* in_sizes, int n_in,
                              void* d_out, int out_size, void* d_ws, size_t ws_size,
                              hipStream_t stream) {
    (void)n_in; (void)out_size; (void)ws_size;
    const float* x  = (const float*)d_in[0];
    const float* y  = (const float*)d_in[1];
    const int*   ei = (const int*)  d_in[2];
    const float* px = (const float*)d_in[3];
    const float* py = (const float*)d_in[4];
    const float* W1 = (const float*)d_in[5];
    const float* b1 = (const float*)d_in[6];
    const float* W2 = (const float*)d_in[7];
    const float* b2 = (const float*)d_in[8];
    float* out = (float*)d_out;

    const int E = in_sizes[2] / 2;               // edge_index is [2, E]
    _Float16* packed = (_Float16*)d_ws;          // 40 KB scratch

    pack_w1_kernel<<<(NFRAG * 512 + 255) / 256, 256, 0, stream>>>(W1, packed);

    const int nblocks = (E + BLOCK_E - 1) / BLOCK_E;
    edge_mlp_kernel<<<nblocks, 128, 0, stream>>>(x, y, ei, px, py, packed,
                                                 b1, W2, b2, out, E);
}
